// AttentionDecoder_63247688401432
// MI455X (gfx1250) — matrix operand order repaired and checked by reading, not compiled, not run on hardware
//
#include <hip/hip_runtime.h>
#include <math.h>

// Problem constants (reference: B=64, S=128, H=512, E=256, V=50000, T=10)
#define BB 64
#define SS 128
#define HH 512
#define EE 256
#define VV 50000
#define TT 10
#define HE 768    /* H + E */
#define H3 1536   /* 3 * H */
#define LOG_EPS -18.420680743952367f

typedef __bf16 bf16;
typedef __attribute__((ext_vector_type(16))) __bf16 v16bf;
typedef __attribute__((ext_vector_type(8)))  __bf16 v8bf;
typedef __attribute__((ext_vector_type(8)))  float  v8f;
typedef __attribute__((ext_vector_type(4)))  unsigned int v4u;

union AFrag { v16bf v; v8bf h[2]; };

// ---------------------------------------------------------------------------
// f32 -> bf16 conversion (grid-stride)
// ---------------------------------------------------------------------------
__global__ void f2bf_kernel(const float* __restrict__ src, bf16* __restrict__ dst, size_t n) {
    size_t stride = (size_t)gridDim.x * blockDim.x;
    for (size_t i = (size_t)blockIdx.x * blockDim.x + threadIdx.x; i < n; i += stride)
        dst[i] = (bf16)src[i];
}

// ---------------------------------------------------------------------------
// Init: SOS step output, h = 0 (fp32 + bf16), idx = 1, sampled[:,0] = 1
// ---------------------------------------------------------------------------
__global__ void init_kernel(float* __restrict__ out, float* __restrict__ hf,
                            bf16* __restrict__ hB, int* __restrict__ idxb,
                            float* __restrict__ sampled) {
    size_t stride = (size_t)gridDim.x * blockDim.x;
    size_t total = (size_t)BB * VV;
    for (size_t i = (size_t)blockIdx.x * blockDim.x + threadIdx.x; i < total; i += stride) {
        size_t b = i / VV, v = i % VV;
        out[b * (size_t)TT * VV + v] = (v == 1) ? 0.0f : LOG_EPS;
        if (i < (size_t)BB * HH) { hf[i] = 0.0f; hB[i] = (bf16)0.0f; }
        if (i < (size_t)BB)      { idxb[i] = 1; sampled[i * TT] = 1.0f; }
    }
}

// ---------------------------------------------------------------------------
// bf16 WMMA GEMM:  C[64][N] = A[64][K] * Bw[N][K]^T + bias[N]
//
// Dataflow tuned for skinny M: the whole A matrix (64xK bf16, 64-96 KB) is
// staged once per block into LDS (padded stride -> bank spread); each wave
// owns one 16-column strip of B/C and accumulates all four 16-row tiles
// (acc[4] = 32 VGPRs), so the B (weight) stream is read exactly once.
// Ragged-N guard uses readfirstlane so it is an SGPR branch: EXEC is never
// modified around v_wmma (ISA requires EXEC all-ones).
// Block = 128 threads (4 waves); grid.x = ceil(N/16 / 4).
// ---------------------------------------------------------------------------
template <int K>
__global__ __launch_bounds__(128)
void gemm_bf16_wmma(const bf16* __restrict__ A,
                    const bf16* __restrict__ Bw,
                    float* __restrict__ C, long long ldc,
                    const float* __restrict__ bias,
                    int N) {
    constexpr int STRIDE = K + 8;                 // +16B: spread LDS banks
    __shared__ __align__(16) bf16 As[64 * STRIDE];

    const int tid  = threadIdx.x;
    const int lane = tid & 31;
    const int half = lane >> 4;                   // lane group 0..15 / 16..31
    const int mn   = lane & 15;                   // row (A) / col (B,C,D)
    const int kb   = half * 16;                   // A fragment K sub-offset

    // Stage A (64 x K bf16) into LDS as 16B chunks. All waves participate.
    constexpr int CHUNKS_PER_ROW = K / 8;         // 8 bf16 = 16 B per chunk
    for (int i = tid; i < 64 * CHUNKS_PER_ROW; i += 128) {
        int r = i / CHUNKS_PER_ROW, c = i % CHUNKS_PER_ROW;
        *reinterpret_cast<v4u*>(&As[r * STRIDE + c * 8]) =
            *reinterpret_cast<const v4u*>(A + (size_t)r * K + c * 8);
    }
    __syncthreads();

    // Wave-uniform column tile id as an SGPR (scalar guard, EXEC untouched).
    const int wave = __builtin_amdgcn_readfirstlane(tid >> 5);
    const int tile = blockIdx.x * 4 + wave;
    const int ntiles = N >> 4;
    if (tile >= ntiles) return;                   // scalar s_cmp/s_cbranch
    const int col0 = tile * 16;

    v8f zero = {0.f, 0.f, 0.f, 0.f, 0.f, 0.f, 0.f, 0.f};
    v8f acc[4];
#pragma unroll
    for (int i = 0; i < 4; ++i) acc[i] = zero;

    // B row for this lane's column: contiguous K bf16.
    const bf16* Brow = Bw + (size_t)(col0 + mn) * K;

    for (int k0 = 0; k0 < K; k0 += 32) {
        // B fragment: lane holds N=mn; K = 16*half .. 16*half+15 contiguous.
        v16bf bm = *reinterpret_cast<const v16bf*>(Brow + k0 + half * 16);
        __builtin_prefetch(Brow + k0 + 256, 0, 3);   // global_prefetch_b8 ahead
#pragma unroll
        for (int mt = 0; mt < 4; ++mt) {
            // A fragment from LDS: lane holds M = mt*16+mn;
            // K = kb..kb+15 contiguous (two ds_load_b128).
            const bf16* arow = &As[(mt * 16 + mn) * STRIDE + k0];
            AFrag a;
            a.h[0] = *reinterpret_cast<const v8bf*>(arow + kb);
            a.h[1] = *reinterpret_cast<const v8bf*>(arow + kb + 8);
            acc[mt] = __builtin_amdgcn_wmma_f32_16x16x32_bf16(
                false, a.v, false, bm, (short)0, acc[mt], false, false);
        }
    }

    const float bia = bias ? bias[col0 + mn] : 0.0f;
#pragma unroll
    for (int mt = 0; mt < 4; ++mt) {
#pragma unroll
        for (int v = 0; v < 8; ++v) {
            // C/D layout: VGPR v -> row v + 8*half; col = mn.
            C[(size_t)(mt * 16 + v + 8 * half) * ldc + col0 + mn] = acc[mt][v] + bia;
        }
    }
}

// ---------------------------------------------------------------------------
// Attention: scores = enc[b] . th[b], softmax over S, ctx = attn @ enc[b],
// then write bf16 x = [embedding[idx[b]], ctx] for the GRU input GEMM.
// grid = B, block = 256 (8 waves; wave per score row).
// ---------------------------------------------------------------------------
__global__ __launch_bounds__(256)
void attn_kernel(const float* __restrict__ enc, const float* __restrict__ th,
                 const float* __restrict__ embedding, const int* __restrict__ idxb,
                 bf16* __restrict__ xB) {
    const int b = blockIdx.x;
    const int tid = threadIdx.x;
    const int wave = tid >> 5;
    const int lane = tid & 31;
    __shared__ float sc[SS];
    __shared__ float red[256];

    const float* encB = enc + (size_t)b * SS * HH;
    const float* thb  = th + (size_t)b * HH;

    // scores
    for (int s = wave; s < SS; s += 8) {
        const float* row = encB + (size_t)s * HH;
        float acc = 0.0f;
        for (int k = lane; k < HH; k += 32) acc += row[k] * thb[k];
#pragma unroll
        for (int off = 16; off > 0; off >>= 1) acc += __shfl_xor(acc, off, 32);
        if (lane == 0) sc[s] = acc;
    }
    __syncthreads();

    // softmax over S=128
    float m = (tid < SS) ? sc[tid] : -INFINITY;
    red[tid] = m; __syncthreads();
    for (int off = 128; off > 0; off >>= 1) {
        if (tid < off) red[tid] = fmaxf(red[tid], red[tid + off]);
        __syncthreads();
    }
    float mx = red[0]; __syncthreads();
    float e = (tid < SS) ? expf(sc[tid] - mx) : 0.0f;
    red[tid] = e; __syncthreads();
    for (int off = 128; off > 0; off >>= 1) {
        if (tid < off) red[tid] += red[tid + off];
        __syncthreads();
    }
    float denom = red[0]; __syncthreads();
    if (tid < SS) sc[tid] = e / denom;
    __syncthreads();

    // ctx -> xB[b][E + h]  (bf16)
    bf16* xrow = xB + (size_t)b * HE;
    for (int h = tid; h < HH; h += 256) {
        float acc = 0.0f;
        for (int s = 0; s < SS; ++s) acc += sc[s] * encB[(size_t)s * HH + h];
        xrow[EE + h] = (bf16)acc;
    }
    // embedding lookup -> xB[b][0..E-1]
    int row = idxb[b];
    if (row > VV) row = 3;
    const float* em = embedding + (size_t)row * EE;
    for (int e2 = tid; e2 < EE; e2 += 256) xrow[e2] = (bf16)em[e2];
}

// ---------------------------------------------------------------------------
// GRU gates: h_new = (1-z)*n + z*h ; writes fp32 h and bf16 h.
// ---------------------------------------------------------------------------
__global__ __launch_bounds__(256)
void gru_gates_kernel(const float* __restrict__ gi, const float* __restrict__ gh,
                      float* __restrict__ hf, bf16* __restrict__ hB) {
    const int b = blockIdx.x;
    const float* gib = gi + (size_t)b * H3;
    const float* ghb = gh + (size_t)b * H3;
    for (int i = threadIdx.x; i < HH; i += 256) {
        float r = 1.0f / (1.0f + expf(-(gib[i] + ghb[i])));
        float z = 1.0f / (1.0f + expf(-(gib[HH + i] + ghb[HH + i])));
        float n = tanhf(gib[2 * HH + i] + r * ghb[2 * HH + i]);
        float hn = (1.0f - z) * n + z * hf[(size_t)b * HH + i];
        hf[(size_t)b * HH + i] = hn;
        hB[(size_t)b * HH + i] = (bf16)hn;
    }
}

// ---------------------------------------------------------------------------
// Per-row (B=64 blocks) max+argmax and sum-exp over V=50000 logits.
// Writes logZ[b], next idx, sampled[b][t] (as float).
// ---------------------------------------------------------------------------
__global__ __launch_bounds__(256)
void logsoftmax_reduce_kernel(const float* __restrict__ out, int t,
                              float* __restrict__ logZ, int* __restrict__ idxb,
                              float* __restrict__ sampled) {
    const int b = blockIdx.x;
    const int tid = threadIdx.x;
    const float* row = out + (size_t)b * TT * VV + (size_t)t * VV;
    __shared__ float rv[256];
    __shared__ int   ri[256];

    float bm = -INFINITY; int bi = 0;
    for (int v = tid; v < VV; v += 256) {
        float x = row[v];
        if (x > bm) { bm = x; bi = v; }
    }
    rv[tid] = bm; ri[tid] = bi; __syncthreads();
    for (int off = 128; off > 0; off >>= 1) {
        if (tid < off) {
            float xo = rv[tid + off]; int io = ri[tid + off];
            if (xo > rv[tid] || (xo == rv[tid] && io < ri[tid])) { rv[tid] = xo; ri[tid] = io; }
        }
        __syncthreads();
    }
    float mx = rv[0]; int amax = ri[0];
    __syncthreads();

    float s = 0.0f;
    for (int v = tid; v < VV; v += 256) s += expf(row[v] - mx);
    rv[tid] = s; __syncthreads();
    for (int off = 128; off > 0; off >>= 1) {
        if (tid < off) rv[tid] += rv[tid + off];
        __syncthreads();
    }
    if (tid == 0) {
        logZ[b] = mx + logf(rv[0]);
        idxb[b] = amax;
        sampled[(size_t)b * TT + t] = (float)amax;
    }
}

__global__ void normalize_kernel(float* __restrict__ out, int t,
                                 const float* __restrict__ logZ) {
    size_t stride = (size_t)gridDim.x * blockDim.x;
    size_t total = (size_t)BB * VV;
    for (size_t i = (size_t)blockIdx.x * blockDim.x + threadIdx.x; i < total; i += stride) {
        int b = (int)(i / VV), v = (int)(i % VV);
        out[(size_t)b * TT * VV + (size_t)t * VV + v] -= logZ[b];
    }
}

// ---------------------------------------------------------------------------
extern "C" void kernel_launch(void* const* d_in, const int* in_sizes, int n_in,
                              void* d_out, int out_size, void* d_ws, size_t ws_size,
                              hipStream_t stream) {
    (void)in_sizes; (void)n_in; (void)out_size; (void)ws_size;
    const float* enc       = (const float*)d_in[0];
    // d_in[1] input_tweets, d_in[2] input_news, d_in[3] final_encoder_hidden: unused
    const float* embedding = (const float*)d_in[4];
    const float* attn_W    = (const float*)d_in[5];
    const float* attn_b    = (const float*)d_in[6];
    const float* W_ih      = (const float*)d_in[7];
    const float* b_ih      = (const float*)d_in[8];
    const float* W_hh      = (const float*)d_in[9];
    const float* b_hh      = (const float*)d_in[10];
    const float* out_W     = (const float*)d_in[11];
    const float* out_b     = (const float*)d_in[12];
    float* out = (float*)d_out;
    float* sampled = out + (size_t)BB * TT * VV;

    // Workspace carve-out (256B aligned)
    char* p = (char*)d_ws;
    auto carve = [&](size_t bytes) { char* r = p; p += (bytes + 255) & ~(size_t)255; return r; };
    bf16* attnWB = (bf16*)carve((size_t)HH * HH * 2);
    bf16* WihB   = (bf16*)carve((size_t)H3 * HE * 2);
    bf16* WhhB   = (bf16*)carve((size_t)H3 * HH * 2);
    bf16* outWB  = (bf16*)carve((size_t)VV * HH * 2);
    float* hf    = (float*)carve((size_t)BB * HH * 4);
    bf16* hB     = (bf16*)carve((size_t)BB * HH * 2);
    bf16* xB     = (bf16*)carve((size_t)BB * HE * 2);
    float* th    = (float*)carve((size_t)BB * HH * 4);
    float* gi    = (float*)carve((size_t)BB * H3 * 4);
    float* gh    = (float*)carve((size_t)BB * H3 * 4);
    float* logZ  = (float*)carve((size_t)BB * 4);
    int*   idxb  = (int*)carve((size_t)BB * 4);

    // One-time per launch: weights -> bf16 (out_W stays hot in the 192MB L2)
    f2bf_kernel<<<512,  256, 0, stream>>>(attn_W, attnWB, (size_t)HH * HH);
    f2bf_kernel<<<1024, 256, 0, stream>>>(W_ih,   WihB,   (size_t)H3 * HE);
    f2bf_kernel<<<1024, 256, 0, stream>>>(W_hh,   WhhB,   (size_t)H3 * HH);
    f2bf_kernel<<<4096, 256, 0, stream>>>(out_W,  outWB,  (size_t)VV * HH);
    init_kernel<<<2048, 256, 0, stream>>>(out, hf, hB, idxb, sampled);

    for (int t = 1; t < TT; ++t) {
        // th = h @ attn_W^T + attn_b        [64 x 512], K=512, 32 tiles
        gemm_bf16_wmma<HH><<<8, 128, 0, stream>>>(
            hB, attnWB, th, HH, attn_b, HH);
        // scores/softmax/ctx + embedding -> x (bf16 [64 x 768])
        attn_kernel<<<BB, 256, 0, stream>>>(enc, th, embedding, idxb, xB);
        // gi = x @ W_ih^T + b_ih            [64 x 1536], K=768, 96 tiles
        gemm_bf16_wmma<HE><<<24, 128, 0, stream>>>(
            xB, WihB, gi, H3, b_ih, H3);
        // gh = h @ W_hh^T + b_hh            [64 x 1536], K=512, 96 tiles
        gemm_bf16_wmma<HH><<<24, 128, 0, stream>>>(
            hB, WhhB, gh, H3, b_hh, H3);
        // gates -> h_new (fp32 + bf16)
        gru_gates_kernel<<<BB, 256, 0, stream>>>(gi, gh, hf, hB);
        // logits = h_new @ out_W^T + out_b  [64 x 50000], 3125 tiles -> 782 blocks
        gemm_bf16_wmma<HH><<<782, 128, 0, stream>>>(
            hB, outWB, out + (size_t)t * VV, (long long)TT * VV, out_b, VV);
        // log-softmax stats + greedy argmax -> next token
        logsoftmax_reduce_kernel<<<BB, 256, 0, stream>>>(out, t, logZ, idxb, sampled);
        normalize_kernel<<<2048, 256, 0, stream>>>(out, t, logZ);
    }
}
